// DConv_40200893890846
// MI455X (gfx1250) — matrix operand order repaired and checked by reading, not compiled, not run on hardware
//
#include <hip/hip_runtime.h>
#include <hip/hip_bf16.h>
#include <stdint.h>

typedef unsigned short u16;
typedef __attribute__((ext_vector_type(16))) __bf16 v16bf;
typedef __attribute__((ext_vector_type(8)))  float  v8f;

union BF16Frag { v16bf v; uint32_t u[8]; };

__device__ __forceinline__ u16 f2bf(float f) {
    uint32_t u = __float_as_uint(f);
    u += 0x7FFFu + ((u >> 16) & 1u);          // round-to-nearest-even
    return (u16)(u >> 16);
}
__device__ __forceinline__ uint32_t f2bf_pk(float lo, float hi) {
    return (uint32_t)f2bf(lo) | ((uint32_t)f2bf(hi) << 16);
}

// A-matrix (16x32 bf16) K index for element-pair ep (0..7) of a lane.
// ISA 7.12.2: lanes 0-15 hold K {0..7,16..23}, lanes 16-31 hold K {8..15,24..31},
// two consecutive K per VGPR.
__device__ __forceinline__ int kpairA(int ep, int hi) {
    return (ep < 4 ? 2 * ep : 2 * ep + 8) + (hi ? 8 : 0);
}
// B-matrix (32x16 bf16): same K order as A (lanes 0-15 K {0..7,16..23}, lanes 16-31 K {8..15,24..31}).
__device__ __forceinline__ int kpairB(int ep, int hi) {
    return (ep < 4 ? 2 * ep : 2 * ep + 8) + (hi ? 8 : 0);
}

#define LDS_STRIDE 264   // 256 + 8 halfword pad -> 528B rows, conflict-free frag reads

// ---------------------------------------------------------------------------
// Prep: fold dw into om weights/bias, build concatenated [value_w ; om_w'] bf16,
// cv1_w bf16, BN scale/bias.
// ---------------------------------------------------------------------------
__global__ __launch_bounds__(256) void k_prep(
    const float* __restrict__ cv1_w, const float* __restrict__ value_w,
    const float* __restrict__ value_b, const float* __restrict__ dw_w,
    const float* __restrict__ dw_b, const float* __restrict__ om_w,
    const float* __restrict__ om_b, const float* __restrict__ gamma,
    const float* __restrict__ beta, const float* __restrict__ mean,
    const float* __restrict__ var,
    u16* __restrict__ W1, u16* __restrict__ Wcat, float* __restrict__ bcat,
    float* __restrict__ sscale, float* __restrict__ sbias)
{
    const int T1 = 65536;          // cv1_w -> bf16
    const int T2 = T1 + 65536;     // value_w -> Wcat rows 0..255
    const int T3 = T2 + 112 * 256; // om_w * dw_w -> Wcat rows 256..367
    const int T4 = T3 + 368;       // bcat
    const int T5 = T4 + 256;       // BN scale/bias
    int stride = gridDim.x * blockDim.x;
    for (int i = blockIdx.x * blockDim.x + threadIdx.x; i < T5; i += stride) {
        if (i < T1) {
            W1[i] = f2bf(cv1_w[i]);
        } else if (i < T2) {
            int j = i - T1;
            Wcat[j] = f2bf(value_w[j]);
        } else if (i < T3) {
            int j = i - T2; int o = j >> 8, c = j & 255;
            Wcat[(256 + o) * 256 + c] = f2bf(om_w[o * 256 + c] * dw_w[c]);
        } else if (i < T4) {
            int j = i - T3;
            if (j < 256) bcat[j] = value_b[j];
            else {
                int o = j - 256;
                float s = om_b[o];
                for (int c = 0; c < 256; ++c) s += om_w[o * 256 + c] * dw_b[c];
                bcat[j] = s;
            }
        } else {
            int j = i - T4;
            float s = gamma[j] * rsqrtf(var[j] + 1e-5f);
            sscale[j] = s;
            sbias[j]  = beta[j] - mean[j] * s;
        }
    }
}

// W3 = cv2_w @ out_w  (folds the last two GEMMs into one), to bf16.
__global__ __launch_bounds__(256) void k_w3(
    const float* __restrict__ cv2_w, const float* __restrict__ out_w,
    u16* __restrict__ W3)
{
    int d2 = blockIdx.x, m = threadIdx.x;
    float s = 0.f;
    for (int d = 0; d < 256; ++d)
        s += cv2_w[d2 * 256 + d] * out_w[d * 256 + m];
    W3[d2 * 256 + m] = f2bf(s);
}

// ---------------------------------------------------------------------------
// Kernel 1: t = cv1(x) via WMMA, kept in LDS (bf16), then fused WMMA against
// concatenated [value_w ; om_w'] -> writes value (NHWC) and offset/mask.
// Block: 256 threads (8 waves), tile = 32 positions of one batch image.
// ---------------------------------------------------------------------------
__global__ __launch_bounds__(256) void k_tvalue(
    const float* __restrict__ x, const u16* __restrict__ W1,
    const u16* __restrict__ Wcat, const float* __restrict__ cv1_b,
    const float* __restrict__ bcat, float* __restrict__ value_ws,
    float* __restrict__ om_ws)
{
    __shared__ u16 xs[32 * LDS_STRIDE];   // x tile, [p][c] bf16
    __shared__ u16 ts[32 * LDS_STRIDE];   // t tile, [p][d] bf16

    const int tid = threadIdx.x;
    const int n  = blockIdx.y;
    const int p0 = blockIdx.x * 32;

    // Cooperative coalesced load of x[n, :, p0:p0+32] -> xs[p][c] (transpose),
    // with global_prefetch a few strided cachelines ahead.
    {
        int p = tid & 31, c0 = tid >> 5;          // c0 in 0..7
        const float* xp = x + (size_t)n * 256 * 4096 + p0 + p;
        #pragma unroll 4
        for (int i = 0; i < 32; ++i) {
            int c = c0 * 32 + i;
            if (i + 4 < 32)
                __builtin_prefetch(xp + (size_t)(c + 4) * 4096, 0, 3);
            xs[p * LDS_STRIDE + c] = f2bf(xp[(size_t)c * 4096]);
        }
    }
    __syncthreads();

    const int wave = tid >> 5, lane = tid & 31;
    const int lm = lane & 15, hi = lane >> 4;

    // Stage 1: t[d,p] = cv1_w @ x ; 16 d-tiles x 2 p-tiles = 32 tiles over 8 waves
    for (int i = 0; i < 4; ++i) {
        int tile = wave * 4 + i;
        int d_t = tile >> 1, p_t = tile & 1;
        v8f acc;
        #pragma unroll
        for (int r = 0; r < 8; ++r) acc[r] = cv1_b[d_t * 16 + r + hi * 8];
        const u16* arow = W1 + (size_t)(d_t * 16 + lm) * 256;
        const u16* brow = &xs[(p_t * 16 + lm) * LDS_STRIDE];
        #pragma unroll
        for (int kt = 0; kt < 8; ++kt) {
            int k0 = kt * 32;
            BF16Frag a, b;
            #pragma unroll
            for (int ep = 0; ep < 8; ++ep) {
                a.u[ep] = *(const uint32_t*)(arow + k0 + kpairA(ep, hi));
                b.u[ep] = *(const uint32_t*)(brow + k0 + kpairB(ep, hi));
            }
            acc = __builtin_amdgcn_wmma_f32_16x16x32_bf16(
                false, a.v, false, b.v, (short)0, acc, false, false);
        }
        // Pack bf16 pairs (consecutive d) -> dword LDS stores
        u16* trow = &ts[(p_t * 16 + lm) * LDS_STRIDE + d_t * 16 + hi * 8];
        #pragma unroll
        for (int r = 0; r < 8; r += 2)
            *(uint32_t*)(trow + r) = f2bf_pk(acc[r], acc[r + 1]);
    }
    __syncthreads();

    // Stage 2: [value ; om] = t @ Wcat^T ; 23 o-tiles x 2 p-tiles = 46 tiles
    for (int tile = wave; tile < 46; tile += 8) {
        int o_t = tile >> 1, p_t = tile & 1;
        v8f acc;
        #pragma unroll
        for (int r = 0; r < 8; ++r) acc[r] = bcat[o_t * 16 + r + hi * 8];
        const u16* arow = Wcat + (size_t)(o_t * 16 + lm) * 256;
        const u16* brow = &ts[(p_t * 16 + lm) * LDS_STRIDE];
        #pragma unroll
        for (int kt = 0; kt < 8; ++kt) {
            int k0 = kt * 32;
            BF16Frag a, b;
            #pragma unroll
            for (int ep = 0; ep < 8; ++ep) {
                a.u[ep] = *(const uint32_t*)(arow + k0 + kpairA(ep, hi));
                b.u[ep] = *(const uint32_t*)(brow + k0 + kpairB(ep, hi));
            }
            acc = __builtin_amdgcn_wmma_f32_16x16x32_bf16(
                false, a.v, false, b.v, (short)0, acc, false, false);
        }
        size_t pg = (size_t)n * 4096 + p0 + p_t * 16 + lm;
        int ob = o_t * 16 + hi * 8;                 // even; tile never straddles 256
        if (ob < 256) {
            float* dst = value_ws + pg * 256 + ob;  // float2 (b64) stores
            #pragma unroll
            for (int r = 0; r < 8; r += 2)
                *(float2*)(dst + r) = make_float2(acc[r], acc[r + 1]);
        } else {
            float* dst = om_ws + pg * 112 + (ob - 256);
            #pragma unroll
            for (int r = 0; r < 8; r += 2)
                *(float2*)(dst + r) = make_float2(acc[r], acc[r + 1]);
        }
    }
}

// ---------------------------------------------------------------------------
// Kernel 2: DCNv4 bilinear sampling -> y tile in LDS (bf16) -> WMMA against
// folded W3 -> BN scale/bias -> SiLU -> NCHW store.
// Block: 256 threads = 32 positions x 4 groups x 2 channel-halves.
// ---------------------------------------------------------------------------
__global__ __launch_bounds__(256) void k_dcn(
    const float* __restrict__ value_ws, const float* __restrict__ om_ws,
    const u16* __restrict__ W3, const float* __restrict__ sscale,
    const float* __restrict__ sbias, float* __restrict__ out)
{
    __shared__ u16 ys[32 * LDS_STRIDE];

    const int tid = threadIdx.x;
    const int n  = blockIdx.y;
    const int p0 = blockIdx.x * 32;

    {
        int p    = tid >> 3;         // 0..31
        int g    = (tid >> 1) & 3;   // group
        int half = tid & 1;          // channel half (2x16 channels each)
        int pp = p0 + p;
        int hh = pp >> 6, ww = pp & 63;

        const float* omp = om_ws + ((size_t)n * 4096 + pp) * 112 + g * 27;

        float fxk[9], fyk[9], mkk[9];
        int   xik[9], yik[9];
        #pragma unroll
        for (int k = 0; k < 9; ++k) {
            float px = (float)ww + (float)(k % 3 - 1) + omp[2 * k];
            float py = (float)hh + (float)(k / 3 - 1) + omp[2 * k + 1];
            float x0 = floorf(px), y0 = floorf(py);
            fxk[k] = px - x0; fyk[k] = py - y0;
            xik[k] = (int)x0; yik[k] = (int)y0;
            mkk[k] = omp[18 + k];   // DCNv4: unnormalized mask, no softmax
        }

        const float* vbase = value_ws + (size_t)n * 4096 * 256 + g * 64;

        for (int c2 = 0; c2 < 2; ++c2) {
            int ch0 = half * 32 + c2 * 16;
            float a[16];
            #pragma unroll
            for (int j = 0; j < 16; ++j) a[j] = 0.f;

            #pragma unroll
            for (int k = 0; k < 9; ++k) {
                float fx = fxk[k], fy = fyk[k], m = mkk[k];
                float w00 = (1.f - fy) * (1.f - fx) * m;
                float w01 = (1.f - fy) * fx * m;
                float w10 = fy * (1.f - fx) * m;
                float w11 = fy * fx * m;
                #pragma unroll
                for (int cy = 0; cy < 2; ++cy) {
                    #pragma unroll
                    for (int cx = 0; cx < 2; ++cx) {
                        float wgt = cy ? (cx ? w11 : w10) : (cx ? w01 : w00);
                        int yy = yik[k] + cy, xx = xik[k] + cx;
                        if (yy >= 0 && yy < 64 && xx >= 0 && xx < 64) {
                            const float4* src = (const float4*)(
                                vbase + ((size_t)(yy * 64 + xx)) * 256 + ch0);
                            #pragma unroll
                            for (int j = 0; j < 4; ++j) {
                                float4 v = src[j];
                                a[4 * j + 0] += wgt * v.x;
                                a[4 * j + 1] += wgt * v.y;
                                a[4 * j + 2] += wgt * v.z;
                                a[4 * j + 3] += wgt * v.w;
                            }
                        }
                    }
                }
            }
            u16* yrow = &ys[p * LDS_STRIDE + g * 64 + ch0];
            #pragma unroll
            for (int j = 0; j < 16; j += 2)
                *(uint32_t*)(yrow + j) = f2bf_pk(a[j], a[j + 1]);
        }
    }
    __syncthreads();

    // out = silu( (y @ W3^T) * s + b ), 16 d-tiles x 2 p-tiles = 32 tiles
    const int wave = tid >> 5, lane = tid & 31;
    const int lm = lane & 15, hi = lane >> 4;
    for (int i = 0; i < 4; ++i) {
        int tile = wave * 4 + i;
        int d_t = tile >> 1, p_t = tile & 1;
        v8f acc = {0.f, 0.f, 0.f, 0.f, 0.f, 0.f, 0.f, 0.f};
        const u16* arow = W3 + (size_t)(d_t * 16 + lm) * 256;
        const u16* brow = &ys[(p_t * 16 + lm) * LDS_STRIDE];
        #pragma unroll
        for (int kt = 0; kt < 8; ++kt) {
            int k0 = kt * 32;
            BF16Frag a, b;
            #pragma unroll
            for (int ep = 0; ep < 8; ++ep) {
                a.u[ep] = *(const uint32_t*)(arow + k0 + kpairA(ep, hi));
                b.u[ep] = *(const uint32_t*)(brow + k0 + kpairB(ep, hi));
            }
            acc = __builtin_amdgcn_wmma_f32_16x16x32_bf16(
                false, a.v, false, b.v, (short)0, acc, false, false);
        }
        #pragma unroll
        for (int r = 0; r < 8; ++r) {
            int d = d_t * 16 + r + hi * 8;
            int p = p_t * 16 + lm;
            float v = acc[r] * sscale[d] + sbias[d];
            float sv = v / (1.f + __expf(-v));          // SiLU
            out[((size_t)n * 256 + d) * 4096 + p0 + p] = sv;
        }
    }
}

// ---------------------------------------------------------------------------
extern "C" void kernel_launch(void* const* d_in, const int* in_sizes, int n_in,
                              void* d_out, int out_size, void* d_ws, size_t ws_size,
                              hipStream_t stream) {
    const float* x       = (const float*)d_in[0];
    const float* cv1_w   = (const float*)d_in[1];
    const float* cv1_b   = (const float*)d_in[2];
    const float* value_w = (const float*)d_in[3];
    const float* value_b = (const float*)d_in[4];
    const float* dw_w    = (const float*)d_in[5];
    const float* dw_b    = (const float*)d_in[6];
    const float* om_w    = (const float*)d_in[7];
    const float* om_b    = (const float*)d_in[8];
    const float* out_w   = (const float*)d_in[9];
    const float* cv2_w   = (const float*)d_in[10];
    const float* gamma   = (const float*)d_in[11];
    const float* beta    = (const float*)d_in[12];
    const float* mean    = (const float*)d_in[13];
    const float* var     = (const float*)d_in[14];
    float* out = (float*)d_out;

    char* ws = (char*)d_ws;
    u16*   W1bf   = (u16*)(ws + 0);            // 131072 B
    u16*   Wcatbf = (u16*)(ws + 131072);       // 188416 B
    u16*   W3bf   = (u16*)(ws + 319488);       // 131072 B
    float* bcat   = (float*)(ws + 450560);     // 1472 B
    float* sscale = (float*)(ws + 452032);     // 1024 B
    float* sbias  = (float*)(ws + 453056);     // 1024 B
    float* val_ws = (float*)(ws + 454144);     // 8*4096*256*4 = 33554432 B
    float* om_ws  = (float*)(ws + 454144 + 33554432ull); // 14680064 B

    k_prep<<<64, 256, 0, stream>>>(cv1_w, value_w, value_b, dw_w, dw_b, om_w,
                                   om_b, gamma, beta, mean, var,
                                   W1bf, Wcatbf, bcat, sscale, sbias);
    k_w3<<<256, 256, 0, stream>>>(cv2_w, out_w, W3bf);
    k_tvalue<<<dim3(128, 8), 256, 0, stream>>>(x, W1bf, Wcatbf, cv1_b, bcat,
                                               val_ws, om_ws);
    k_dcn<<<dim3(128, 8), 256, 0, stream>>>(val_ws, om_ws, W3bf, sscale, sbias,
                                            out);
}
